// Processor_61710090109114
// MI455X (gfx1250) — matrix operand order repaired and checked by reading, not compiled, not run on hardware
//
#include <hip/hip_runtime.h>

#define DLAT 128
#define NSTEP 2
#define NM_ 30000
#define NO_ 2000
#define EMM_ 240000
#define EMO_ 60000

typedef _Float16 half_t;
typedef __attribute__((ext_vector_type(16))) _Float16 v16h;
typedef __attribute__((ext_vector_type(8)))  _Float16 v8h;
typedef __attribute__((ext_vector_type(8)))  float    v8f;
typedef __attribute__((ext_vector_type(4)))  float    v4f;

static __device__ __forceinline__ v8f vzero8() {
  v8f z = {0.f, 0.f, 0.f, 0.f, 0.f, 0.f, 0.f, 0.f};
  return z;
}

// A fragment (16x32 f16, MxK) from row-major f16 rows in LDS.
// ISA layout: lane l holds row (l&15); half h=l>>4: K = h*8+j (j<8), 16+h*8+(j-8) (j>=8).
static __device__ __forceinline__ v16h load_a16(const half_t* base, int pitch,
                                                int row0, int k0, int lane) {
  const int r = lane & 15, hh = lane >> 4;
  const half_t* p = base + (size_t)(row0 + r) * pitch + (k0 + hh * 8);
  union { v16h v; v8h h[2]; } u;
  u.h[0] = *(const v8h*)(p);
  u.h[1] = *(const v8h*)(p + 16);
  return u.v;
}

// B fragment (32x16 f16, KxN) from W^T stored row-major [N=128][K].
// ISA layout: lane l holds col (l&15); half h=l>>4: K = h*8+j (j<8), 16+h*8+(j-8) (j>=8).
static __device__ __forceinline__ v16h load_b16(const half_t* Wt, int K,
                                                int nt, int k0, int lane) {
  const int n = nt * 16 + (lane & 15), hh = lane >> 4;
  const half_t* p = Wt + (size_t)n * K + (k0 + hh * 8);
  union { v16h v; v8h h[2]; } u;
  u.h[0] = *(const v8h*)(p);
  u.h[1] = *(const v8h*)(p + 16);
  return u.v;
}

// 16xK @ KxN(=128) GEMM tile for one wave, software-pipelined:
// double-buffered A/B fragment registers so the next chunk's loads (clause of
// 16 global_load_b128 + 2 ds_load_b128) overlap the current chunk's 8 WMMAs.
template <int K>
static __device__ __forceinline__ void gemm16(const half_t* aLds, int row0,
                                              const half_t* Wt, int lane, v8f acc[8]) {
  constexpr int NCH = K / 32;
  v16h a_cur, b_cur[8], a_nxt, b_nxt[8];
  a_cur = load_a16(aLds, K, row0, 0, lane);
#pragma unroll
  for (int nt = 0; nt < 8; ++nt) b_cur[nt] = load_b16(Wt, K, nt, 0, lane);
#pragma unroll
  for (int ch = 0; ch < NCH; ++ch) {
    if (ch + 1 < NCH) {
      const int k_nxt = (ch + 1) * 32;
      a_nxt = load_a16(aLds, K, row0, k_nxt, lane);
#pragma unroll
      for (int nt = 0; nt < 8; ++nt) b_nxt[nt] = load_b16(Wt, K, nt, k_nxt, lane);
    }
#pragma unroll
    for (int nt = 0; nt < 8; ++nt)
      acc[nt] = __builtin_amdgcn_wmma_f32_16x16x32_f16(
          false, a_cur, false, b_cur[nt], (short)0, acc[nt], false, false);
    if (ch + 1 < NCH) {
      a_cur = a_nxt;
#pragma unroll
      for (int nt = 0; nt < 8; ++nt) b_cur[nt] = b_nxt[nt];
    }
  }
}

// One fused MLP+LayerNorm conv layer.
// Edge mode:  row = edge e; input = [x_dst[ei_dst[e]] | x_src[ei_src[e]] | ea[e]] (K0=384)
//             out[e] = ea[e] + msg;   atomicAdd(agg[ei_dst[e]], msg)
// Node mode:  row = node i; input = [x[i] | agg[i]] (K0=256);  out[i] = x[i] + msg
template <int K0, bool IS_EDGE>
__global__ __launch_bounds__(128) void mlp_ln_kernel(
    const float* __restrict__ xA,      // edge: x_dst ; node: x
    const float* __restrict__ xB,      // edge: x_src ; node: agg
    const int*   __restrict__ ei_src,
    const int*   __restrict__ ei_dst,
    const float* __restrict__ res_in,  // edge: ea_in ; node: x
    float*       __restrict__ out,
    float*       __restrict__ agg_out, // edge only
    int nrows,
    const half_t* __restrict__ W0t, const float* __restrict__ b0,
    const half_t* __restrict__ W1t, const float* __restrict__ b1,
    const half_t* __restrict__ W2t, const float* __restrict__ b2,
    const float* __restrict__ gw, const float* __restrict__ bw) {
  extern __shared__ char smem[];
  half_t* act0 = (half_t*)smem;                          // 64 x K0 f16
  half_t* hstA = (half_t*)(smem + (size_t)64 * K0 * 2);  // 4 waves x 16x128 f16
  half_t* hstB = hstA + 4 * 16 * 128;                    // 4 waves x 16x128 f16
  float*  lnS  = (float*)(hstB + 4 * 16 * 128);          // 4 waves x 16x128 f32
  float*  stats = lnS + 4 * 16 * 128;                    // 4 waves x 32 f32

  const int tid = threadIdx.x;
  const int lane = tid & 31;
  const int wave = tid >> 5;
  const int row_base = blockIdx.x * 64;

  // ---- phase 1: gather + f32->f16 pack into LDS rows ----
  const int SEG = K0 / 4;  // float4 segments per row
  for (int i = tid; i < 64 * SEG; i += 128) {
    const int r = i / SEG, q = i - r * SEG;
    const int e = row_base + r;
    const int ec = (e < nrows) ? e : 0;
    const float* src;
    if (IS_EDGE) {
      if (q < 32)      src = xA + (size_t)ei_dst[ec] * DLAT + q * 4;
      else if (q < 64) src = xB + (size_t)ei_src[ec] * DLAT + (q - 32) * 4;
      else             src = res_in + (size_t)ec * DLAT + (q - 64) * 4;
    } else {
      if (q < 32)      src = xA + (size_t)ec * DLAT + q * 4;
      else             src = xB + (size_t)ec * DLAT + (q - 32) * 4;
    }
    const v4f v = *(const v4f*)src;
    half_t* d = act0 + (size_t)r * K0 + q * 4;
    d[0] = (half_t)v[0]; d[1] = (half_t)v[1];
    d[2] = (half_t)v[2]; d[3] = (half_t)v[3];
  }
  __syncthreads();

  const int wrow0 = wave * 16;
  const int c0 = lane & 15, hh = lane >> 4;
  v8f acc[8];

  // ---- layer 0: K0 -> 128, bias + relu ----
#pragma unroll
  for (int nt = 0; nt < 8; ++nt) acc[nt] = vzero8();
  gemm16<K0>(act0, wrow0, W0t, lane, acc);
  half_t* hA = hstA + wave * (16 * 128);
#pragma unroll
  for (int nt = 0; nt < 8; ++nt) {
    const int col = nt * 16 + c0;
    const float bb = b0[col];
#pragma unroll
    for (int g8 = 0; g8 < 8; ++g8) {
      const float v = acc[nt][g8] + bb;
      hA[(g8 + 8 * hh) * 128 + col] = (half_t)(v > 0.f ? v : 0.f);
    }
  }
  __syncthreads();

  // ---- layer 1: 128 -> 128, bias + relu ----
#pragma unroll
  for (int nt = 0; nt < 8; ++nt) acc[nt] = vzero8();
  gemm16<128>(hA, 0, W1t, lane, acc);
  half_t* hB = hstB + wave * (16 * 128);
#pragma unroll
  for (int nt = 0; nt < 8; ++nt) {
    const int col = nt * 16 + c0;
    const float bb = b1[col];
#pragma unroll
    for (int g8 = 0; g8 < 8; ++g8) {
      const float v = acc[nt][g8] + bb;
      hB[(g8 + 8 * hh) * 128 + col] = (half_t)(v > 0.f ? v : 0.f);
    }
  }
  __syncthreads();

  // ---- layer 2: 128 -> 128, bias, stage f32 for LayerNorm ----
#pragma unroll
  for (int nt = 0; nt < 8; ++nt) acc[nt] = vzero8();
  gemm16<128>(hB, 0, W2t, lane, acc);
  float* ln = lnS + wave * (16 * 128);
#pragma unroll
  for (int nt = 0; nt < 8; ++nt) {
    const int col = nt * 16 + c0;
    const float bb = b2[col];
#pragma unroll
    for (int g8 = 0; g8 < 8; ++g8)
      ln[(g8 + 8 * hh) * 128 + col] = acc[nt][g8] + bb;
  }
  __syncthreads();

  // ---- LayerNorm stats: lanes 0..15 each own one row ----
  float* st = stats + wave * 32;
  if (lane < 16) {
    const float* rp = ln + lane * 128;
    float s = 0.f;
    for (int c = 0; c < 128; c += 4) {
      const v4f x = *(const v4f*)(rp + c);
      s += x[0] + x[1] + x[2] + x[3];
    }
    const float mean = s * 0.0078125f;
    float vs = 0.f;
    for (int c = 0; c < 128; c += 4) {
      const v4f x = *(const v4f*)(rp + c);
      const float d0 = x[0] - mean, d1 = x[1] - mean;
      const float d2 = x[2] - mean, d3 = x[3] - mean;
      vs += d0 * d0 + d1 * d1 + d2 * d2 + d3 * d3;
    }
    st[lane * 2] = mean;
    st[lane * 2 + 1] = rsqrtf(vs * 0.0078125f + 1e-5f);
  }
  __syncthreads();

  // ---- normalize + residual + aggregation (coalesced, 32 lanes x 4 chunks) ----
  for (int r = 0; r < 16; ++r) {
    const int e = row_base + wrow0 + r;
    const bool ok = e < nrows;
    const float mean = st[r * 2], rstd = st[r * 2 + 1];
    int dst = 0;
    if (IS_EDGE) dst = ok ? ei_dst[e] : 0;
#pragma unroll
    for (int k = 0; k < 4; ++k) {
      const int c = k * 32 + lane;
      const float val = (ln[r * 128 + c] - mean) * rstd * gw[c] + bw[c];
      if (ok) {
        const size_t idx = (size_t)e * DLAT + c;
        out[idx] = res_in[idx] + val;
        if (IS_EDGE) atomicAdd(agg_out + (size_t)dst * DLAT + c, val);
      }
    }
  }
}

// W: [m][K][128] f32  ->  Wt: [m][128][K] f16
__global__ void transpose_to_f16(const float* __restrict__ W, half_t* __restrict__ Wt,
                                 int K, long long total) {
  const long long i = (long long)blockIdx.x * blockDim.x + threadIdx.x;
  if (i >= total) return;
  const long long kn = (long long)K * 128;
  const long long m = i / kn;
  const long long rem = i - m * kn;
  const int k = (int)(rem / 128);
  const int n = (int)(rem - (long long)k * 128);
  Wt[m * kn + (long long)n * K + k] = (half_t)W[i];
}

extern "C" void kernel_launch(void* const* d_in, const int* in_sizes, int n_in,
                              void* d_out, int out_size, void* d_ws, size_t ws_size,
                              hipStream_t stream) {
  (void)in_sizes; (void)n_in; (void)out_size; (void)ws_size;
  const float* x_mesh = (const float*)d_in[0];
  const float* x_obj  = (const float*)d_in[1];
  const int*   ei_mm  = (const int*)d_in[2];   // [2][EMM]: row0=src, row1=dst
  const int*   ei_mo  = (const int*)d_in[3];   // [2][EMO]
  const float* ea_mm  = (const float*)d_in[4];
  const float* ea_mo  = (const float*)d_in[5];
  const float* eW0 = (const float*)d_in[6];  const float* eb0 = (const float*)d_in[7];
  const float* eW1 = (const float*)d_in[8];  const float* eb1 = (const float*)d_in[9];
  const float* eW2 = (const float*)d_in[10]; const float* eb2 = (const float*)d_in[11];
  const float* eg  = (const float*)d_in[12]; const float* ebt = (const float*)d_in[13];
  const float* nW0 = (const float*)d_in[14]; const float* nb0 = (const float*)d_in[15];
  const float* nW1 = (const float*)d_in[16]; const float* nb1 = (const float*)d_in[17];
  const float* nW2 = (const float*)d_in[18]; const float* nb2 = (const float*)d_in[19];
  const float* ng  = (const float*)d_in[20]; const float* nbt = (const float*)d_in[21];

  // ---- carve workspace ----
  char* w = (char*)d_ws;
  auto carve = [&](size_t bytes) -> char* {
    char* p = w;
    w += (bytes + 255) & ~(size_t)255;
    return p;
  };
  half_t* eW0t = (half_t*)carve((size_t)4 * 384 * 128 * 2);
  half_t* eW1t = (half_t*)carve((size_t)4 * 128 * 128 * 2);
  half_t* eW2t = (half_t*)carve((size_t)4 * 128 * 128 * 2);
  half_t* nW0t = (half_t*)carve((size_t)4 * 256 * 128 * 2);
  half_t* nW1t = (half_t*)carve((size_t)4 * 128 * 128 * 2);
  half_t* nW2t = (half_t*)carve((size_t)4 * 128 * 128 * 2);
  float* xm_ws   = (float*)carve((size_t)NM_ * 128 * 4);
  float* xo_ws   = (float*)carve((size_t)NO_ * 128 * 4);
  float* eamm_ws = (float*)carve((size_t)EMM_ * 128 * 4);
  float* eamo_ws = (float*)carve((size_t)EMO_ * 128 * 4);
  float* agg     = (float*)carve((size_t)(NM_ + NO_) * 128 * 4);
  float* agg_obj = agg + (size_t)NM_ * 128;

  // ---- prep: transpose + convert all weight stacks (2 types x 2 steps) ----
  auto tr = [&](const float* W, half_t* Wt, int K) {
    const long long total = 4LL * K * 128;
    transpose_to_f16<<<dim3((unsigned)((total + 255) / 256)), dim3(256), 0, stream>>>(W, Wt, K, total);
  };
  tr(eW0, eW0t, 384); tr(eW1, eW1t, 128); tr(eW2, eW2t, 128);
  tr(nW0, nW0t, 256); tr(nW1, nW1t, 128); tr(nW2, nW2t, 128);

  float* xm_final = (float*)d_out;
  float* xo_final = (float*)d_out + (size_t)NM_ * 128;

  const size_t SH_EDGE = (size_t)64 * 384 * 2 + 2 * (4 * 16 * 128 * 2) + 4 * 16 * 128 * 4 + 4 * 32 * 4;
  const size_t SH_NODE = (size_t)64 * 256 * 2 + 2 * (4 * 16 * 128 * 2) + 4 * 16 * 128 * 4 + 4 * 32 * 4;

  const float* xm = x_mesh;
  const float* xo = x_obj;
  const float* ea0 = ea_mm;
  const float* ea1 = ea_mo;

  for (int s = 0; s < NSTEP; ++s) {
    hipMemsetAsync(agg, 0, (size_t)(NM_ + NO_) * 128 * 4, stream);

    {  // edge conv mesh->mesh (type 0)
      const int idx = 0 * NSTEP + s;
      mlp_ln_kernel<384, true><<<(EMM_ + 63) / 64, 128, SH_EDGE, stream>>>(
          xm, xm, ei_mm, ei_mm + EMM_, ea0, eamm_ws, agg, EMM_,
          eW0t + (size_t)idx * 384 * 128, eb0 + idx * 128,
          eW1t + (size_t)idx * 128 * 128, eb1 + idx * 128,
          eW2t + (size_t)idx * 128 * 128, eb2 + idx * 128,
          eg + idx * 128, ebt + idx * 128);
    }
    {  // edge conv mesh->obj (type 1): dst = obj nodes, src = mesh nodes
      const int idx = 1 * NSTEP + s;
      mlp_ln_kernel<384, true><<<(EMO_ + 63) / 64, 128, SH_EDGE, stream>>>(
          xo, xm, ei_mo, ei_mo + EMO_, ea1, eamo_ws, agg_obj, EMO_,
          eW0t + (size_t)idx * 384 * 128, eb0 + idx * 128,
          eW1t + (size_t)idx * 128 * 128, eb1 + idx * 128,
          eW2t + (size_t)idx * 128 * 128, eb2 + idx * 128,
          eg + idx * 128, ebt + idx * 128);
    }
    float* xm_next = (s == NSTEP - 1) ? xm_final : xm_ws;
    float* xo_next = (s == NSTEP - 1) ? xo_final : xo_ws;
    {  // node conv mesh (type 0)
      const int idx = 0 * NSTEP + s;
      mlp_ln_kernel<256, false><<<(NM_ + 63) / 64, 128, SH_NODE, stream>>>(
          xm, agg, nullptr, nullptr, xm, xm_next, nullptr, NM_,
          nW0t + (size_t)idx * 256 * 128, nb0 + idx * 128,
          nW1t + (size_t)idx * 128 * 128, nb1 + idx * 128,
          nW2t + (size_t)idx * 128 * 128, nb2 + idx * 128,
          ng + idx * 128, nbt + idx * 128);
    }
    {  // node conv obj (type 1)
      const int idx = 1 * NSTEP + s;
      mlp_ln_kernel<256, false><<<(NO_ + 63) / 64, 128, SH_NODE, stream>>>(
          xo, agg_obj, nullptr, nullptr, xo, xo_next, nullptr, NO_,
          nW0t + (size_t)idx * 256 * 128, nb0 + idx * 128,
          nW1t + (size_t)idx * 128 * 128, nb1 + idx * 128,
          nW2t + (size_t)idx * 128 * 128, nb2 + idx * 128,
          ng + idx * 128, nbt + idx * 128);
    }
    xm = xm_next; xo = xo_next;
    ea0 = eamm_ws; ea1 = eamo_ws;
  }
}
